// Phi_78082505441622
// MI455X (gfx1250) — matrix operand order repaired and checked by reading, not compiled, not run on hardware
//
#include <hip/hip_runtime.h>
#include <hip/hip_bf16.h>
#include <math.h>

typedef _Float16 h16;
typedef __attribute__((ext_vector_type(16))) _Float16 v16h;
typedef __attribute__((ext_vector_type(8)))  float    v8f;
typedef unsigned int v4u __attribute__((ext_vector_type(4)));
typedef int          v8i_t __attribute__((ext_vector_type(8)));
typedef int          v4i_t __attribute__((ext_vector_type(4)));

#define D1    64      // d+1
#define MM    256     // hidden width m
#define NEX   4096
#define TILE  16      // examples per forward block
#define HSTEP 0.5f    // h = 1/(nTh-1)

#define LOG2E_X2 2.885390082f   // 2*log2(e)
#define LN2      0.69314718056f

// ---------------- fast transcendentals (CDNA5 TRANS unit, branch-free) ----------------
__device__ __forceinline__ float fast_tanh(float t) {
#if __has_builtin(__builtin_amdgcn_tanhf)
  return __builtin_amdgcn_tanhf(t);          // v_tanh_f32
#else
  float a = fabsf(t);
  float e = __builtin_amdgcn_exp2f(-LOG2E_X2 * a);   // exp(-2a)
  float r = (1.0f - e) / (1.0f + e);
  return __builtin_copysignf(r, t);
#endif
}

// antideriv_tanh(t) = |t| + log1p(exp(-2|t|))
__device__ __forceinline__ float antideriv_tanh(float t) {
  float a = fabsf(t);
  float e = __builtin_amdgcn_exp2f(-LOG2E_X2 * a);   // in (0,1], no overflow
  return a + LN2 * __builtin_amdgcn_logf(1.0f + e);  // v_log_f32 (log2)
}

// ---------------- WMMA fragment helpers (wave32, 16x16x32 f16) ----------------
// A (16x32): lane l -> row = l&15 ; kbase = (l&16)?8:0 ; elem e<8 -> k0+kb+e, e>=8 -> k0+16+kb+(e-8)
__device__ __forceinline__ v16h load_a_frag(const h16* __restrict__ base, int rowStride, int k0) {
  const int l  = threadIdx.x & 31;
  const h16* p = base + (size_t)(l & 15) * rowStride + k0 + ((l & 16) ? 8 : 0);
  v16h a;
#pragma unroll
  for (int e = 0; e < 8; ++e) a[e] = p[e];
#pragma unroll
  for (int e = 0; e < 8; ++e) a[8 + e] = p[16 + e];
  return a;
}

// B (32x16): lane l -> col = l&15 ; kbase = (l&16)?8:0 ; elem e<8 -> k0+kb+e, e>=8 -> k0+16+kb+(e-8)
// source stored column-by-column: src[col*colStride + k] (K contiguous)
__device__ __forceinline__ v16h load_b_frag(const h16* __restrict__ base, int colStride, int k0) {
  const int l  = threadIdx.x & 31;
  const h16* p = base + (size_t)(l & 15) * colStride + k0 + ((l & 16) ? 8 : 0);
  v16h b;
#pragma unroll
  for (int e = 0; e < 8; ++e) b[e] = p[e];
#pragma unroll
  for (int e = 0; e < 8; ++e) b[8 + e] = p[16 + e];
  return b;
}

__device__ __forceinline__ v8f wmma_f16f32(v16h a, v16h b, v8f c) {
  return __builtin_amdgcn_wmma_f32_16x16x32_f16(false, a, false, b, (short)0, c, false, false);
}

// ---------------- TDM: 1D tensor copy global -> LDS (D# per ISA 8.3/8.4) ----------------
#if __has_builtin(__builtin_amdgcn_tensor_load_to_lds) && __has_builtin(__builtin_amdgcn_s_wait_tensorcnt)
#define HAVE_TDM 1
__device__ __forceinline__ void tdm_copy_1d(const void* gsrc, void* ldst,
                                            unsigned elems, unsigned dsz_code) {
  unsigned long long ga = (unsigned long long)(uintptr_t)gsrc;
  unsigned lds = (unsigned)(uintptr_t)ldst;   // addrspace(3) byte offset
  v4u g0;
  g0[0] = 1u;                                           // count=1, user descriptor
  g0[1] = lds;                                          // lds_addr [63:32]
  g0[2] = (unsigned)(ga & 0xffffffffu);                 // global_addr [95:64]
  g0[3] = (unsigned)((ga >> 32) & 0x1ffffffu) | (2u << 30); // addr[56:32] | type=2
  v8i_t g1;
  g1[0] = (int)(dsz_code << 16);                        // wg_mask=0, data_size
  g1[1] = (int)((elems & 0xffffu) << 16);               // tensor_dim0[15:0]
  g1[2] = (int)(((elems >> 16) & 0xffffu) | (1u << 16));// tensor_dim0 hi | tensor_dim1=1
  g1[3] = (int)((elems & 0xffffu) << 16);               // tile_dim0 = elems
  g1[4] = 1;                                            // tile_dim1=1, tile_dim2=0
  g1[5] = (int)elems;                                   // tensor_dim0_stride lo
  g1[6] = 0;                                            // stride hi, dim1_stride lo
  g1[7] = 0;
  v4i_t z4 = {0, 0, 0, 0};
  v8i_t z8 = {0, 0, 0, 0, 0, 0, 0, 0};
  // 6-arg form on this toolchain: (g0, g1, g2, g3, g4, cpol)
  __builtin_amdgcn_tensor_load_to_lds(g0, g1, z4, z4, z8, 0);
}
#else
#define HAVE_TDM 0
#endif

// ---------------- prep: f16 weight copies (+transposes), symA, row sumsq, trace ----------------
__global__ __launch_bounds__(256) void phi_prep_kernel(
    const float* __restrict__ W0, const float* __restrict__ Wd, const float* __restrict__ A,
    h16* __restrict__ W0h, h16* __restrict__ W0Th,
    h16* __restrict__ Wd0h, h16* __restrict__ Wd1h,
    h16* __restrict__ Wd0Th, h16* __restrict__ Wd1Th,
    float* __restrict__ symA, float* __restrict__ rssq, float* __restrict__ trace)
{
  const int gid = blockIdx.x * blockDim.x + threadIdx.x;
  const int gsz = gridDim.x * blockDim.x;
  for (int i = gid; i < MM * D1; i += gsz) W0h[i] = (h16)W0[i];
  for (int i = gid; i < D1 * MM; i += gsz) { int r = i >> 8, k = i & 255; W0Th[i] = (h16)W0[k * D1 + r]; }
  for (int i = gid; i < MM * MM; i += gsz) {
    Wd0h[i] = (h16)Wd[i];
    Wd1h[i] = (h16)Wd[MM * MM + i];
    int m = i >> 8, k = i & 255;
    Wd0Th[i] = (h16)Wd[k * MM + m];
    Wd1Th[i] = (h16)Wd[MM * MM + k * MM + m];
  }
  for (int i = gid; i < D1 * D1; i += gsz) {
    int r = i >> 6, c = i & 63;
    float s = 0.f;
    for (int j = 0; j < 10; ++j) s += A[j * D1 + r] * A[j * D1 + c];
    symA[i] = s;
  }
  for (int i = gid; i < MM; i += gsz) {
    float s = 0.f;
    for (int c = 0; c < 63; ++c) { float w = W0[i * D1 + c]; s += w * w; }
    rssq[i] = s;
  }
  if (gid == 0) {
    float s = 0.f;
    for (int i = 0; i < 63; ++i)
      for (int j = 0; j < 10; ++j) { float a = A[j * D1 + i]; s += a * a; }
    *trace = s;
  }
}

// ---------------- forward+backward-z: grad and trH base ----------------
__global__ __launch_bounds__(256) void phi_forward_kernel(
    const float* __restrict__ x, const float* __restrict__ b0, const float* __restrict__ bd,
    const float* __restrict__ wv, const float* __restrict__ cw,
    const h16* __restrict__ W0h, const h16* __restrict__ W0Th,
    const h16* __restrict__ Wd0h, const h16* __restrict__ Wd1h,
    const h16* __restrict__ Wd0Th, const h16* __restrict__ Wd1Th,
    const float* __restrict__ symA, const float* __restrict__ rssq, const float* __restrict__ trace,
    h16* __restrict__ g_to, h16* __restrict__ g_tp1,
    float* __restrict__ g_w1, float* __restrict__ g_w2,
    float* __restrict__ out)
{
  __shared__ h16   sXh[TILE * D1];     // x tile, f16, [ex][k]
  __shared__ h16   sTO[TILE * MM];     // tanh(opening), [ex][m]
  __shared__ h16   sTP1[TILE * MM];    // tanh(pre1), [ex][m]
  __shared__ float sFeat[TILE * MM];   // running feature, f32
  __shared__ h16   sFeatH[TILE * MM];  // running feature, f16 (GEMM A)
  __shared__ h16   sS[TILE * MM];      // staged B operand [ex][k=m]
  __shared__ float sZ0[TILE * D1];
  __shared__ float sTrh[TILE];

  const int tid    = threadIdx.x;
  const int w      = tid >> 5;
  const int l      = tid & 31;
  const int colL   = l & 15;
  const int rowOff = (l & 16) ? 8 : 0;
  const int exg0   = blockIdx.x * TILE;

  if (tid < TILE) sTrh[tid] = 0.f;
  for (int i = tid; i < TILE * D1; i += 256) sXh[i] = (h16)x[(size_t)exg0 * D1 + i];
  __syncthreads();

  // ---- opening = x @ W0^T + b0 : M=16ex, N=256, K=64 ----
  for (int half = 0; half < 2; ++half) {
    const int nb = (w + half * 8) * 16;
    v8f c;
    { float b = b0[nb + colL];
#pragma unroll
      for (int q = 0; q < 8; ++q) c[q] = b; }
#pragma unroll
    for (int ks = 0; ks < 2; ++ks) {
      v16h a = load_a_frag(sXh, D1, ks * 32);
      v16h b = load_b_frag(W0h + (size_t)nb * D1, D1, ks * 32);
      c = wmma_f16f32(a, b, c);
    }
#pragma unroll
    for (int q = 0; q < 8; ++q) {
      int ex = rowOff + q, n = nb + colL;
      float op = c[q], th = fast_tanh(op);
      sTO[ex * MM + n] = (h16)th;
      g_to[(size_t)(exg0 + ex) * MM + n] = (h16)th;
      float u = antideriv_tanh(op);
      sFeat[ex * MM + n] = u;
      sFeatH[ex * MM + n] = (h16)u;
    }
  }
  __syncthreads();

  // ---- pre1 = feat @ Wd0^T + bd0 (two-phase: reads & writes sFeatH) ----
  v8f p1[2];
  for (int half = 0; half < 2; ++half) {
    const int nb = (w + half * 8) * 16;
    v8f c;
    { float b = bd[nb + colL];
#pragma unroll
      for (int q = 0; q < 8; ++q) c[q] = b; }
    for (int ks = 0; ks < 8; ++ks) {
      v16h a = load_a_frag(sFeatH, MM, ks * 32);
      v16h b = load_b_frag(Wd0h + (size_t)nb * MM, MM, ks * 32);
      c = wmma_f16f32(a, b, c);
    }
    p1[half] = c;
  }
  __syncthreads();
  for (int half = 0; half < 2; ++half) {
    const int nb = (w + half * 8) * 16;
#pragma unroll
    for (int q = 0; q < 8; ++q) {
      int ex = rowOff + q, n = nb + colL;
      float p = p1[half][q], th = fast_tanh(p);
      sTP1[ex * MM + n] = (h16)th;
      g_tp1[(size_t)(exg0 + ex) * MM + n] = (h16)th;
      float f = sFeat[ex * MM + n] + HSTEP * antideriv_tanh(p);
      sFeat[ex * MM + n] = f;
      sFeatH[ex * MM + n] = (h16)f;
    }
  }
  __syncthreads();

  // ---- pre2 = feat @ Wd1^T + bd1 ; emit s2' = h*tanh(pre2)*wv, wgt2 ----
  for (int half = 0; half < 2; ++half) {
    const int nb = (w + half * 8) * 16;
    v8f c;
    { float b = bd[MM + nb + colL];
#pragma unroll
      for (int q = 0; q < 8; ++q) c[q] = b; }
    for (int ks = 0; ks < 8; ++ks) {
      v16h a = load_a_frag(sFeatH, MM, ks * 32);
      v16h b = load_b_frag(Wd1h + (size_t)nb * MM, MM, ks * 32);
      c = wmma_f16f32(a, b, c);
    }
#pragma unroll
    for (int q = 0; q < 8; ++q) {
      int ex = rowOff + q, n = nb + colL;
      float p = c[q], th = fast_tanh(p);
      float wvn = wv[n];
      sS[ex * MM + n] = (h16)(HSTEP * th * wvn);
      g_w2[(size_t)(exg0 + ex) * MM + n] = (1.0f - th * th) * wvn;
    }
  }
  __syncthreads();

  // ---- z2 = wv + Wd1^T @ s2'  (rows=m, cols=ex; wave owns m-tiles 2w,2w+1) ----
  v8f zacc[2];
  for (int j = 0; j < 2; ++j) {
    const int mb = (2 * w + j) * 16;
    v8f c;
#pragma unroll
    for (int q = 0; q < 8; ++q) c[q] = wv[mb + rowOff + q];
    for (int ks = 0; ks < 8; ++ks) {
      v16h a = load_a_frag(Wd1Th + (size_t)mb * MM, MM, ks * 32);
      v16h b = load_b_frag(sS, MM, ks * 32);
      c = wmma_f16f32(a, b, c);
    }
    zacc[j] = c;
  }
  __syncthreads();   // all reads of sS(s2') done
  for (int j = 0; j < 2; ++j) {
    const int mb = (2 * w + j) * 16;
#pragma unroll
    for (int q = 0; q < 8; ++q) {
      int m = mb + rowOff + q, ex = colL;
      float z2v = zacc[j][q];
      float tp = (float)sTP1[ex * MM + m];
      sS[ex * MM + m] = (h16)(HSTEP * tp * z2v);                 // s1'
      g_w1[(size_t)(exg0 + ex) * MM + m] = (1.0f - tp * tp) * z2v;
    }
  }
  __syncthreads();

  // ---- z1 = z2 + Wd0^T @ s1'  (accumulate onto z2 regs) ----
  for (int j = 0; j < 2; ++j) {
    const int mb = (2 * w + j) * 16;
    v8f c = zacc[j];
    for (int ks = 0; ks < 8; ++ks) {
      v16h a = load_a_frag(Wd0Th + (size_t)mb * MM, MM, ks * 32);
      v16h b = load_b_frag(sS, MM, ks * 32);
      c = wmma_f16f32(a, b, c);
    }
    zacc[j] = c;
  }
  __syncthreads();   // all reads of sS(s1') done
  {
    float partial = 0.f;
    for (int j = 0; j < 2; ++j) {
      const int mb = (2 * w + j) * 16;
#pragma unroll
      for (int q = 0; q < 8; ++q) {
        int m = mb + rowOff + q, ex = colL;
        float z1v = zacc[j][q];
        float to = (float)sTO[ex * MM + m];
        sS[ex * MM + m] = (h16)(to * z1v);                       // s0
        partial += (1.0f - to * to) * z1v * rssq[m];
      }
    }
    atomicAdd(&sTrh[colL], partial);
  }
  __syncthreads();

  // ---- z0 = W0^T @ s0 : 64 rows -> 4 m-tiles (waves 0..3) ----
  if (w < 4) {
    const int db = w * 16;
    v8f c;
#pragma unroll
    for (int q = 0; q < 8; ++q) c[q] = 0.f;
    for (int ks = 0; ks < 8; ++ks) {
      v16h a = load_a_frag(W0Th + (size_t)db * MM, MM, ks * 32);
      v16h b = load_b_frag(sS, MM, ks * 32);
      c = wmma_f16f32(a, b, c);
    }
#pragma unroll
    for (int q = 0; q < 8; ++q)
      sZ0[colL * D1 + db + rowOff + q] = c[q];
  }
  __syncthreads();

  // ---- grad = z0 + symA @ x^T + cw ; write grad.T row-major ----
  for (int idx = tid; idx < TILE * D1; idx += 256) {
    int ex = idx >> 6, r = idx & 63;
    float acc = sZ0[ex * D1 + r] + cw[r];
    const float* xr = x + (size_t)(exg0 + ex) * D1;
    for (int cc = 0; cc < D1; ++cc) acc += symA[r * D1 + cc] * xr[cc];
    out[(size_t)(exg0 + ex) * D1 + r] = acc;
  }
  if (tid < TILE)
    out[(size_t)NEX * D1 + exg0 + tid] = sTrh[tid] + trace[0];
}

// ---------------- Jacobian propagation: trH += h * sum(wgt * (Wd @ Jac)^2) ----------------
__global__ __launch_bounds__(256) void phi_jac_kernel(
    const float* __restrict__ W0,
    const h16* __restrict__ Wd0h, const h16* __restrict__ Wd1h,
    const h16* __restrict__ g_to, const h16* __restrict__ g_tp1,
    const float* __restrict__ g_w1, const float* __restrict__ g_w2,
    float* __restrict__ trH)
{
  __shared__ h16   sJac[64 * MM];   // [dd][k=m], K contiguous for B fragments (32 KB)
  __shared__ h16   sTO[MM];         // per-example staged vectors (TDM)
  __shared__ h16   sTP1[MM];
  __shared__ float sW1[MM];
  __shared__ float sW2[MM];

  const int n = blockIdx.x;
  const int tid = threadIdx.x, w = tid >> 5, l = tid & 31;
  const int colL = l & 15, rowOff = (l & 16) ? 8 : 0;
  const size_t base = (size_t)n * MM;

#if HAVE_TDM
  // Tensor Data Mover: async DMA of per-example vectors into LDS (TENSORcnt)
  if (tid < 32) {
    tdm_copy_1d(g_to  + base, sTO,  MM, 1u);  // 2-byte elements
    tdm_copy_1d(g_tp1 + base, sTP1, MM, 1u);
    tdm_copy_1d(g_w1  + base, sW1,  MM, 2u);  // 4-byte elements
    tdm_copy_1d(g_w2  + base, sW2,  MM, 2u);
    __builtin_amdgcn_s_wait_tensorcnt(0);
  }
#else
  for (int i = tid; i < MM; i += 256) {
    sTO[i]  = g_to[base + i];
    sTP1[i] = g_tp1[base + i];
    sW1[i]  = g_w1[base + i];
    sW2[i]  = g_w2[base + i];
  }
#endif
  __syncthreads();

  // Jac[m][dd] = Kopen[m][dd] * tanhopen[m] ; pad column 63 with zeros
  for (int i = tid; i < 64 * MM; i += 256) {
    int dd = i >> 8, m = i & 255;
    float v = (dd < 63) ? W0[m * D1 + dd] * (float)sTO[m] : 0.f;
    sJac[i] = (h16)v;
  }
  __syncthreads();

  float tpart = 0.f;
  v8f acc[8];

  // ---- layer 1: KJ = Wd0 @ Jac ; t1 += wgt1*KJ^2 ; Jac += h*tanh(pre1)*KJ ----
  for (int t = 0; t < 8; ++t) {
    const int tile = w * 8 + t, mt = tile >> 2, nt = tile & 3;
    const h16* Abase = Wd0h + (size_t)(mt * 16) * MM;
    __builtin_prefetch((const void*)(Abase + (size_t)colL * MM), 0, 1);
    v8f c;
#pragma unroll
    for (int q = 0; q < 8; ++q) c[q] = 0.f;
    for (int ks = 0; ks < 8; ++ks) {
      v16h a = load_a_frag(Abase, MM, ks * 32);
      v16h b = load_b_frag(sJac + (size_t)(nt * 16) * MM, MM, ks * 32);
      c = wmma_f16f32(a, b, c);
    }
    acc[t] = c;
#pragma unroll
    for (int q = 0; q < 8; ++q) {
      int m = mt * 16 + rowOff + q;
      float kv = c[q];
      tpart += sW1[m] * kv * kv;
    }
  }
  __syncthreads();   // all B reads of sJac complete
  for (int t = 0; t < 8; ++t) {
    const int tile = w * 8 + t, mt = tile >> 2, nt = tile & 3;
    const int dd = nt * 16 + colL;
#pragma unroll
    for (int q = 0; q < 8; ++q) {
      int m = mt * 16 + rowOff + q;
      float tp = (float)sTP1[m];
      int idx = dd * MM + m;
      sJac[idx] = (h16)((float)sJac[idx] + HSTEP * tp * acc[t][q]);
    }
  }
  __syncthreads();

  // ---- layer 2: KJ2 = Wd1 @ Jac ; t2 += wgt2*KJ2^2 ----
  for (int t = 0; t < 8; ++t) {
    const int tile = w * 8 + t, mt = tile >> 2, nt = tile & 3;
    const h16* Abase = Wd1h + (size_t)(mt * 16) * MM;
    __builtin_prefetch((const void*)(Abase + (size_t)colL * MM), 0, 1);
    v8f c;
#pragma unroll
    for (int q = 0; q < 8; ++q) c[q] = 0.f;
    for (int ks = 0; ks < 8; ++ks) {
      v16h a = load_a_frag(Abase, MM, ks * 32);
      v16h b = load_b_frag(sJac + (size_t)(nt * 16) * MM, MM, ks * 32);
      c = wmma_f16f32(a, b, c);
    }
#pragma unroll
    for (int q = 0; q < 8; ++q) {
      int m = mt * 16 + rowOff + q;
      float kv = c[q];
      tpart += sW2[m] * kv * kv;
    }
  }

  // wave-level reduction (all lanes share example n), one atomic per wave
  for (int off = 16; off > 0; off >>= 1) tpart += __shfl_down(tpart, off, 32);
  if (l == 0) atomicAdd(&trH[n], HSTEP * tpart);
}

// ---------------- launch ----------------
extern "C" void kernel_launch(void* const* d_in, const int* in_sizes, int n_in,
                              void* d_out, int out_size, void* d_ws, size_t ws_size,
                              hipStream_t stream) {
  const float* x  = (const float*)d_in[0];
  const float* W0 = (const float*)d_in[1];
  const float* b0 = (const float*)d_in[2];
  const float* Wd = (const float*)d_in[3];
  const float* bd = (const float*)d_in[4];
  const float* wv = (const float*)d_in[5];
  const float* A  = (const float*)d_in[6];
  const float* cw = (const float*)d_in[7];
  float* out = (float*)d_out;

  char* ws = (char*)d_ws;
  size_t off = 0;
  auto alloc = [&](size_t bytes) -> void* {
    void* p = ws + off;
    off += (bytes + 255) & ~(size_t)255;
    return p;
  };
  h16*   W0h   = (h16*)alloc((size_t)MM * D1 * 2);
  h16*   W0Th  = (h16*)alloc((size_t)D1 * MM * 2);
  h16*   Wd0h  = (h16*)alloc((size_t)MM * MM * 2);
  h16*   Wd1h  = (h16*)alloc((size_t)MM * MM * 2);
  h16*   Wd0Th = (h16*)alloc((size_t)MM * MM * 2);
  h16*   Wd1Th = (h16*)alloc((size_t)MM * MM * 2);
  float* symA  = (float*)alloc((size_t)D1 * D1 * 4);
  float* rssq  = (float*)alloc((size_t)MM * 4);
  float* trace = (float*)alloc(4);
  h16*   g_to  = (h16*)alloc((size_t)NEX * MM * 2);
  h16*   g_tp1 = (h16*)alloc((size_t)NEX * MM * 2);
  float* g_w1  = (float*)alloc((size_t)NEX * MM * 4);
  float* g_w2  = (float*)alloc((size_t)NEX * MM * 4);

  phi_prep_kernel<<<64, 256, 0, stream>>>(W0, Wd, A, W0h, W0Th, Wd0h, Wd1h, Wd0Th, Wd1Th,
                                          symA, rssq, trace);
  phi_forward_kernel<<<NEX / TILE, 256, 0, stream>>>(x, b0, bd, wv, cw,
                                                     W0h, W0Th, Wd0h, Wd1h, Wd0Th, Wd1Th,
                                                     symA, rssq, trace,
                                                     g_to, g_tp1, g_w1, g_w2, out);
  phi_jac_kernel<<<NEX, 256, 0, stream>>>(W0, Wd0h, Wd1h, g_to, g_tp1, g_w1, g_w2,
                                          out + (size_t)NEX * D1);
}
